// RotaryMultiheadAttention_83090437308530
// MI455X (gfx1250) — matrix operand order repaired and checked by reading, not compiled, not run on hardware
//
#include <hip/hip_runtime.h>
#include <hip/hip_bf16.h>

// ---------------------------------------------------------------------------
// Rotary multihead attention, CDNA5 (gfx1250, wave32, WMMA bf16 16x16x32)
//   B=4, L=2048, E=1024, H=16, D=64
// Pipeline:
//   0) cvt_kernel      : one-shot fp32->bf16 for activations and weights.
//   1) qkv_rope_kernel : Q/K/V = x @ W^T + b, fused RoPE (Q also *1/sqrt(D));
//                        Q,K -> [B,H,L,D] bf16 ; V -> [B,H,D,L] bf16
//   2) attn_kernel     : flash-style online softmax over ALL keys (additive
//                        -1e9 causal + padding masks = exact reference
//                        semantics). K/V tiles staged in LDS via gfx1250
//                        GLOBAL_LOAD_ASYNC_TO_LDS_B128 (ASYNCcnt), waited
//                        with s_wait_asynccnt before the workgroup barrier;
//                        next tile prefetched with global_prefetch_b8.
//   3) oproj_kernel    : out = O @ wo^T + bo  (fp32 output)
// Workspace: 104 MB bf16 (attn output aliases the consumed Xq buffer).
// ---------------------------------------------------------------------------

#define BB 4
#define LL 2048
#define EE 1024
#define HH 16
#define DD 64
#define NEG_INF (-1e9f)

#if defined(__AMDGCN__) && __has_builtin(__builtin_amdgcn_global_load_async_to_lds_b128)
#define HAS_ASYNC_LDS 1
#endif

typedef __attribute__((ext_vector_type(8)))  __bf16 v8bf;
typedef __attribute__((ext_vector_type(16))) __bf16 v16bf;
typedef __attribute__((ext_vector_type(8)))  float  v8f;
typedef __attribute__((ext_vector_type(4)))  int    v4i;

static __device__ inline v8bf load_bf8(const __bf16* p) {
    return *(const v8bf*)p;                 // 16B load (global_load_b128 / ds_load_b128)
}
static __device__ inline v8bf cvt_bf8(const float* p) {
    v8bf o;
#pragma unroll
    for (int i = 0; i < 8; ++i) o[i] = (__bf16)p[i];
    return o;
}
static __device__ inline v16bf cat16(v8bf a, v8bf b) {
    return __builtin_shufflevector(a, b, 0,1,2,3,4,5,6,7,8,9,10,11,12,13,14,15);
}
static __device__ inline v8f wmma_bf16(v16bf a, v16bf b, v8f c) {
    // D = A(16x32) * B(32x16) + C, f32 accumulate
    return __builtin_amdgcn_wmma_f32_16x16x32_bf16(false, a, false, b, (short)0, c, false, false);
}
static __device__ inline v8f zero_v8f() {
    v8f z;
#pragma unroll
    for (int i = 0; i < 8; ++i) z[i] = 0.0f;
    return z;
}

// --- gfx1250 async global->LDS staging (16B per lane) ----------------------
static __device__ inline void stage_b128(__bf16* lds_dst, const __bf16* gsrc) {
#if defined(HAS_ASYNC_LDS)
    __builtin_amdgcn_global_load_async_to_lds_b128(
        (__attribute__((address_space(1))) v4i*)gsrc,
        (__attribute__((address_space(3))) v4i*)lds_dst,
        /*imm offset*/0, /*cpol*/0);
#else
    *(v8bf*)lds_dst = load_bf8(gsrc);
#endif
}
static __device__ inline void async_wait0() {
#if defined(HAS_ASYNC_LDS)
#if __has_builtin(__builtin_amdgcn_s_wait_asynccnt)
    __builtin_amdgcn_s_wait_asynccnt(0);
#else
    asm volatile("s_wait_asynccnt 0x0" ::: "memory");
#endif
#endif
}

// ---------------------------------------------------------------------------
// Kernel 0: bulk fp32 -> bf16 conversion (n must be a multiple of 8).
// ---------------------------------------------------------------------------
__global__ __launch_bounds__(256)
void cvt_kernel(const float* __restrict__ src, __bf16* __restrict__ dst, int n)
{
    const int i = (blockIdx.x * 256 + threadIdx.x) * 8;
    if (i < n) *(v8bf*)(dst + i) = cvt_bf8(src + i);
}

// ---------------------------------------------------------------------------
// Kernel 1: QKV projection + bias + RoPE (all-bf16 operands, pure WMMA loop).
// grid = (M/128, H, 3) ; block = 256 (8 waves, each wave: 16 rows x 64 cols)
// ---------------------------------------------------------------------------
__global__ __launch_bounds__(256)
void qkv_rope_kernel(const __bf16* __restrict__ xq, const __bf16* __restrict__ xk,
                     const __bf16* __restrict__ xv,
                     const __bf16* __restrict__ wq, const __bf16* __restrict__ wk,
                     const __bf16* __restrict__ wv,
                     const float* __restrict__ bq, const float* __restrict__ bk,
                     const float* __restrict__ bv,
                     __bf16* __restrict__ outQ, __bf16* __restrict__ outK,
                     __bf16* __restrict__ outV)
{
    const int mode = blockIdx.z;                       // 0=Q 1=K 2=V
    const __bf16* X  = (mode == 0) ? xq : (mode == 1) ? xk : xv;
    const __bf16* W  = (mode == 0) ? wq : (mode == 1) ? wk : wv;
    const float*  Bi = (mode == 0) ? bq : (mode == 1) ? bk : bv;
    __bf16*       Out = (mode == 0) ? outQ : (mode == 1) ? outK : outV;

    const int lane = threadIdx.x & 31;
    const int wave = threadIdx.x >> 5;
    const int nl   = lane & 15;          // D-matrix col / A-matrix row / B col
    const int hb   = lane >> 4;          // half-wave select
    const int hi8  = hb * 8;             // D-matrix row offset
    const int kb   = hb * 8;             // A-matrix K interleave base
    const int rowbase = blockIdx.x * 128 + wave * 16;
    const int h = blockIdx.y;            // N tile == one head (64 cols)

    v8f acc[4];
#pragma unroll
    for (int j = 0; j < 4; ++j) acc[j] = zero_v8f();

    const __bf16* arow = X + (size_t)(rowbase + nl) * EE;

    for (int k0 = 0; k0 < EE; k0 += 32) {
        // A fragment, ISA 16-bit A 16x32 layout
        v16bf af = cat16(load_bf8(arow + k0 + kb), load_bf8(arow + k0 + 16 + kb));
#pragma unroll
        for (int j = 0; j < 4; ++j) {
            const int n = h * 64 + j * 16 + nl;        // W row (B[k][n] = W[n][k])
            const __bf16* wrow = W + (size_t)n * EE + k0 + kb;
            v16bf bf = cat16(load_bf8(wrow), load_bf8(wrow + 16));
            acc[j] = wmma_bf16(af, bf, acc[j]);
        }
    }

    // bias
#pragma unroll
    for (int j = 0; j < 4; ++j) {
        const float bb = Bi[h * 64 + j * 16 + nl];
#pragma unroll
        for (int r = 0; r < 8; ++r) acc[j][r] += bb;
    }

    const int bidx = rowbase >> 11;                    // rowbase / L

    if (mode < 2) {
        // RoPE: pair (d, d+32), cos/sin(l * 10000^(-d/32)); Q also * 1/sqrt(D)
        const float scale = (mode == 0) ? 0.125f : 1.0f;
#pragma unroll
        for (int j = 0; j < 2; ++j) {
            const int dd = j * 16 + nl;                // 0..31
            const float invf = __powf(10000.0f, -(float)dd * (1.0f / 32.0f));
#pragma unroll
            for (int r = 0; r < 8; ++r) {
                const int gm = rowbase + r + hi8;
                const int l  = gm & (LL - 1);
                float c, s;
                __sincosf((float)l * invf, &c, &s);
                const float x1 = acc[j][r], x2 = acc[j + 2][r];
                const float o1 = (x1 * c - x2 * s) * scale;
                const float o2 = (x2 * c + x1 * s) * scale;
                const size_t base = ((size_t)(bidx * HH + h) * LL + l) * DD;
                Out[base + dd]      = (__bf16)o1;
                Out[base + dd + 32] = (__bf16)o2;
            }
        }
    } else {
        // V: no rope, store transposed [B,H,D,L]
#pragma unroll
        for (int j = 0; j < 4; ++j) {
            const int d = j * 16 + nl;
#pragma unroll
            for (int r = 0; r < 8; ++r) {
                const int gm = rowbase + r + hi8;
                const int l  = gm & (LL - 1);
                Out[((size_t)(bidx * HH + h) * DD + d) * LL + l] = (__bf16)acc[j][r];
            }
        }
    }
}

// ---------------------------------------------------------------------------
// Kernel 2: attention. grid = (L/128, H, B), block = 256 (8 waves).
// Each wave owns 16 query rows; the block stages each 32-key K tile (32x64)
// and V tile (64x32) into LDS once per iteration with async-to-LDS loads.
// ---------------------------------------------------------------------------
__global__ __launch_bounds__(256)
void attn_kernel(const __bf16* __restrict__ Q, const __bf16* __restrict__ K,
                 const __bf16* __restrict__ V,
                 const unsigned char* __restrict__ kpm,
                 __bf16* __restrict__ O)
{
    // [0,2048)  : K tile, [pos][d]   32x64, stride 64 halves
    // [2048,4096): V tile, [d][pos]  64x32, stride 32 halves
    // [4096,...) : per-wave 16x32 P tiles, stride 40 halves
    __shared__ __bf16 smem[4096 + 8 * 16 * 40];
    __bf16* kT = smem;
    __bf16* vT = smem + 2048;

    const int tid  = threadIdx.x;
    const int lane = tid & 31;
    const int wave = tid >> 5;
    const int nl   = lane & 15;
    const int hb   = lane >> 4;
    const int hi8  = hb * 8;
    const int kb   = hb * 8;

    const int b = blockIdx.z, h = blockIdx.y;
    const int qb = blockIdx.x * 128 + wave * 16;

    const __bf16* Qh = Q + (size_t)(b * HH + h) * LL * DD;
    const __bf16* Kh = K + (size_t)(b * HH + h) * LL * DD;
    const __bf16* Vh = V + (size_t)(b * HH + h) * DD * LL;
    const unsigned char* pm = kpm + (size_t)b * LL;

    // cooperative-staging coordinates (one b128 per thread per tile)
    const int krow = tid >> 3, kseg = (tid & 7) * 8;   // K: 32 rows x 8 segs
    const int vrow = tid >> 2, vseg = (tid & 3) * 8;   // V: 64 rows x 4 segs

    // Q A-fragments (row m = nl), d 0..31 and 32..63 (already *1/sqrt(D))
    const __bf16* qp = Qh + (size_t)(qb + nl) * DD;
    const v16bf qlo = cat16(load_bf8(qp + kb),      load_bf8(qp + 16 + kb));
    const v16bf qhi = cat16(load_bf8(qp + 32 + kb), load_bf8(qp + 48 + kb));

    v8f accO[4];
#pragma unroll
    for (int j = 0; j < 4; ++j) accO[j] = zero_v8f();
    float mrun[8], lsum[8];
#pragma unroll
    for (int r = 0; r < 8; ++r) { mrun[r] = -1e30f; lsum[r] = 0.0f; }

    __bf16* pT = smem + 4096 + wave * 640;

    for (int kv = 0; kv < LL; kv += 32) {
        // ---- stage K (32x64) and V (64x32) tiles into LDS (async) ---------
        stage_b128(&kT[krow * 64 + kseg], Kh + (size_t)(kv + krow) * DD + kseg);
        stage_b128(&vT[vrow * 32 + vseg], Vh + (size_t)vrow * LL + kv + vseg);
        if (kv + 32 < LL) {           // pull next tiles toward the WGP
            __builtin_prefetch(Kh + (size_t)(kv + 32 + krow) * DD + kseg, 0, 3);
            __builtin_prefetch(Vh + (size_t)vrow * LL + kv + 32 + vseg, 0, 3);
        }
        async_wait0();                // ASYNCcnt==0: tile landed in LDS
        __syncthreads();

        // ---- S = Q K^T for two 16-key tiles (K frags from LDS) ------------
        v8f S[2];
#pragma unroll
        for (int t = 0; t < 2; ++t) {
            const int pr = (t * 16 + nl) * 64;         // LDS row of key position
            const v16bf k0 = cat16(load_bf8(&kT[pr + kb]),
                                   load_bf8(&kT[pr + 16 + kb]));       // d 0..31
            const v16bf k1 = cat16(load_bf8(&kT[pr + 32 + kb]),
                                   load_bf8(&kT[pr + 48 + kb]));  // d 32..63
            v8f s = wmma_bf16(qlo, k0, zero_v8f());
            S[t]  = wmma_bf16(qhi, k1, s);
        }
        // ---- additive causal + key-padding masks --------------------------
#pragma unroll
        for (int t = 0; t < 2; ++t) {
            const int keyg = kv + t * 16 + nl;
            const float padd = pm[keyg] ? NEG_INF : 0.0f;
#pragma unroll
            for (int r = 0; r < 8; ++r) {
                const int qg = qb + r + hi8;
                S[t][r] += padd + ((keyg > qg) ? NEG_INF : 0.0f);
            }
        }
        // ---- online softmax update; write P (bf16) into per-wave LDS ------
#pragma unroll
        for (int r = 0; r < 8; ++r) {
            const float s0 = S[0][r], s1 = S[1][r];
            float rv = fmaxf(s0, s1);
#pragma unroll
            for (int off = 8; off > 0; off >>= 1) rv = fmaxf(rv, __shfl_xor(rv, off, 16));
            const float nm = fmaxf(mrun[r], rv);
            const float al = __expf(mrun[r] - nm);
            const float p0 = __expf(s0 - nm);
            const float p1 = __expf(s1 - nm);
            float rs = p0 + p1;
#pragma unroll
            for (int off = 8; off > 0; off >>= 1) rs += __shfl_xor(rs, off, 16);
            lsum[r] = lsum[r] * al + rs;
            mrun[r] = nm;
#pragma unroll
            for (int j = 0; j < 4; ++j) accO[j][r] *= al;
            pT[(r + hi8) * 40 + nl]      = (__bf16)p0;
            pT[(r + hi8) * 40 + 16 + nl] = (__bf16)p1;
        }
        // per-wave LDS RAW: DS ops from one wave execute in order — no barrier
        const v16bf pf = cat16(load_bf8(&pT[nl * 40 + kb]),
                               load_bf8(&pT[nl * 40 + 16 + kb]));
        // ---- O += P V (V frags from LDS) ----------------------------------
#pragma unroll
        for (int j = 0; j < 4; ++j) {
            const int vr = (j * 16 + nl) * 32;
            const v16bf vf = cat16(load_bf8(&vT[vr + kb]),
                                   load_bf8(&vT[vr + 16 + kb]));
            accO[j] = wmma_bf16(pf, vf, accO[j]);
        }
        __syncthreads();   // protect K/V tiles before next staging round
    }

    // ---- epilogue: divide by row sums, store O in [B, L, E] bf16 ----------
#pragma unroll
    for (int r = 0; r < 8; ++r) {
        const float inv = 1.0f / lsum[r];
        const int l = qb + r + hi8;
        const size_t base = ((size_t)b * LL + l) * EE + h * DD;
#pragma unroll
        for (int j = 0; j < 4; ++j)
            O[base + j * 16 + nl] = (__bf16)(accO[j][r] * inv);
    }
}

// ---------------------------------------------------------------------------
// Kernel 3: output projection (bf16 x bf16 -> fp32). grid = (M/128, E/64).
// ---------------------------------------------------------------------------
__global__ __launch_bounds__(256)
void oproj_kernel(const __bf16* __restrict__ A, const __bf16* __restrict__ W,
                  const float* __restrict__ Bi, float* __restrict__ out)
{
    const int lane = threadIdx.x & 31;
    const int wave = threadIdx.x >> 5;
    const int nl   = lane & 15;
    const int hb   = lane >> 4;
    const int hi8  = hb * 8;
    const int kb   = hb * 8;
    const int rowbase = blockIdx.x * 128 + wave * 16;
    const int ncol0   = blockIdx.y * 64;

    v8f acc[4];
#pragma unroll
    for (int j = 0; j < 4; ++j) acc[j] = zero_v8f();

    const __bf16* arow = A + (size_t)(rowbase + nl) * EE;

    for (int k0 = 0; k0 < EE; k0 += 32) {
        v16bf af = cat16(load_bf8(arow + k0 + kb), load_bf8(arow + k0 + 16 + kb));
#pragma unroll
        for (int j = 0; j < 4; ++j) {
            const int n = ncol0 + j * 16 + nl;
            const __bf16* wrow = W + (size_t)n * EE + k0 + kb;
            v16bf bf = cat16(load_bf8(wrow), load_bf8(wrow + 16));
            acc[j] = wmma_bf16(af, bf, acc[j]);
        }
    }
#pragma unroll
    for (int j = 0; j < 4; ++j) {
        const int n = ncol0 + j * 16 + nl;
        const float bb = Bi[n];
#pragma unroll
        for (int r = 0; r < 8; ++r) {
            const int gm = rowbase + r + hi8;
            out[(size_t)gm * EE + n] = acc[j][r] + bb;
        }
    }
}

// ---------------------------------------------------------------------------
extern "C" void kernel_launch(void* const* d_in, const int* in_sizes, int n_in,
                              void* d_out, int out_size, void* d_ws, size_t ws_size,
                              hipStream_t stream)
{
    const float* query = (const float*)d_in[0];
    const float* key   = (const float*)d_in[1];
    const float* value = (const float*)d_in[2];
    const float* wq    = (const float*)d_in[3];
    const float* bq    = (const float*)d_in[4];
    const float* wk    = (const float*)d_in[5];
    const float* bk    = (const float*)d_in[6];
    const float* wv    = (const float*)d_in[7];
    const float* bv    = (const float*)d_in[8];
    const float* wo    = (const float*)d_in[9];
    const float* bo    = (const float*)d_in[10];
    const unsigned char* kpm = (const unsigned char*)d_in[11];  // bool mask [B,L]
    float* out = (float*)d_out;

    const size_t nAct = (size_t)BB * LL * EE;   // 8,388,608 elements
    const size_t nW   = (size_t)EE * EE;        // 1,048,576 elements

    __bf16* Xq = (__bf16*)d_ws;                 // bf16 workspace layout (elements):
    __bf16* Xk = Xq + nAct;
    __bf16* Xv = Xk + nAct;
    __bf16* Wq = Xv + nAct;
    __bf16* Wk = Wq + nW;
    __bf16* Wv = Wk + nW;
    __bf16* Wo = Wv + nW;
    __bf16* Qw = Wo + nW;                       // [B,H,L,D]
    __bf16* Kw = Qw + nAct;                     // [B,H,L,D]
    __bf16* Vw = Kw + nAct;                     // [B,H,D,L]
    __bf16* Ob = Xq;                            // alias: Xq fully consumed before attn
    (void)ws_size; (void)in_sizes; (void)n_in; (void)out_size;  // 104 MB used

    dim3 blk(256);
    // 0) one-shot fp32 -> bf16
    cvt_kernel<<<dim3(nAct / 8 / 256), blk, 0, stream>>>(query, Xq, (int)nAct);
    cvt_kernel<<<dim3(nAct / 8 / 256), blk, 0, stream>>>(key,   Xk, (int)nAct);
    cvt_kernel<<<dim3(nAct / 8 / 256), blk, 0, stream>>>(value, Xv, (int)nAct);
    cvt_kernel<<<dim3(nW / 8 / 256),   blk, 0, stream>>>(wq, Wq, (int)nW);
    cvt_kernel<<<dim3(nW / 8 / 256),   blk, 0, stream>>>(wk, Wk, (int)nW);
    cvt_kernel<<<dim3(nW / 8 / 256),   blk, 0, stream>>>(wv, Wv, (int)nW);
    cvt_kernel<<<dim3(nW / 8 / 256),   blk, 0, stream>>>(wo, Wo, (int)nW);
    // 1..3) projections, attention, output projection
    qkv_rope_kernel<<<dim3((BB * LL) / 128, HH, 3), blk, 0, stream>>>(
        Xq, Xk, Xv, Wq, Wk, Wv, bq, bk, bv, Qw, Kw, Vw);
    attn_kernel<<<dim3(LL / 128, HH, BB), blk, 0, stream>>>(Qw, Kw, Vw, kpm, Ob);
    oproj_kernel<<<dim3((BB * LL) / 128, EE / 64), blk, 0, stream>>>(Ob, Wo, bo, out);
}
